// SlidingWindowAttention_20736102105398
// MI455X (gfx1250) — matrix operand order repaired and checked by reading, not compiled, not run on hardware
//
#include <hip/hip_runtime.h>

typedef __attribute__((ext_vector_type(16))) __bf16 bf16x16;
typedef __attribute__((ext_vector_type(8)))  __bf16 bf16x8;
typedef __attribute__((ext_vector_type(4)))  __bf16 bf16x4;
typedef __attribute__((ext_vector_type(8)))  float  f32x8;

constexpr int Bc = 2, Tc = 2048, Cc = 1024, Hc = 16, HD = 64;

// ---------------------------------------------------------------------------
// CDNA5 async global->LDS copy (16B per lane), tracked by ASYNCcnt.
// vdst = LDS byte address, vaddr = 32-bit byte offset, saddr = 64-bit base.
// ---------------------------------------------------------------------------
static __device__ __forceinline__ void async_copy16(unsigned lds_addr,
                                                    unsigned long long sbase,
                                                    unsigned voff) {
  asm volatile("global_load_async_to_lds_b128 %0, %1, %2"
               :: "v"(lds_addr), "v"(voff), "s"(sbase)
               : "memory");
}
static __device__ __forceinline__ unsigned lds_off(const void* p) {
  return (unsigned)(unsigned long long)p;  // low 32 bits = LDS byte offset
}

// ---------------------------------------------------------------------------
// Fragment loaders (CDNA5 ISA 7.12.2, 16-bit A 16x32 / B 32x16)
// A: lane half=0 -> K{0..7,16..23}; half=1 -> K{8..15,24..31}; row = lane%16
// B: lane half=0 -> K{0..7,16..23}; half=1 -> K{8..15,24..31}; column = lane%16
// ---------------------------------------------------------------------------
static __device__ __forceinline__ bf16x16 frag_ld_a(const __bf16* row, int half) {
  bf16x8 a = *reinterpret_cast<const bf16x8*>(row + half * 8);
  bf16x8 b = *reinterpret_cast<const bf16x8*>(row + half * 8 + 16);
  bf16x16 r;
#pragma unroll
  for (int i = 0; i < 8; ++i) { r[i] = a[i]; r[i + 8] = b[i]; }
  return r;
}
static __device__ __forceinline__ bf16x16 frag_ld_b(const __bf16* row, int half) {
  bf16x8 a = *reinterpret_cast<const bf16x8*>(row + half * 8);
  bf16x8 b = *reinterpret_cast<const bf16x8*>(row + half * 8 + 16);
  bf16x16 r;
#pragma unroll
  for (int i = 0; i < 8; ++i) { r[i] = a[i]; r[i + 8] = b[i]; }
  return r;
}

// ---------------------------------------------------------------------------
// One-shot precision/layout conversion kernels (bandwidth-trivial).
// ---------------------------------------------------------------------------
__global__ __launch_bounds__(256) void cvt_f32_bf16_kernel(
    const float* __restrict__ s, __bf16* __restrict__ d, int n4) {
  int i = blockIdx.x * 256 + threadIdx.x;
  if (i < n4) {
    float4 v = reinterpret_cast<const float4*>(s)[i];
    bf16x4 o;
    o[0] = (__bf16)v.x; o[1] = (__bf16)v.y;
    o[2] = (__bf16)v.z; o[3] = (__bf16)v.w;
    *reinterpret_cast<bf16x4*>(d + (size_t)i * 4) = o;
  }
}
// dst[n][k] = (bf16) src[k][n], K fixed at 1024 (coalesced writes)
__global__ __launch_bounds__(256) void cvt_transpose_kernel(
    const float* __restrict__ s, __bf16* __restrict__ d, int Ndim, int total) {
  int i = blockIdx.x * 256 + threadIdx.x;
  if (i < total) {
    int n = i >> 10, k = i & 1023;
    d[i] = (__bf16)s[(size_t)k * Ndim + n];
  }
}

// ---------------------------------------------------------------------------
// GEMM: C[M,N] = A[M,K] * Bt[N,K]^T, bf16 WMMA, f32 accumulate.
// Async double-buffered LDS staging (3 async b128 per thread per tile).
// MODE 0: N=3072, epilogue scatters q/k/v bf16 ws [B][H][T][hd]
// MODE 1: N=1024, epilogue -> f32 d_out
// Block: 256 threads (8 waves), tile 128(M) x 64(N) x 32(K), wave tile 32x32.
// ---------------------------------------------------------------------------
constexpr int BM = 128, BN = 64, BK = 32;
constexpr int LDA = BK + 16;  // 48 el -> 96B rows (16B aligned, conflict pad)
constexpr int LDB = BK + 16;

template <int MODE>
__global__ __launch_bounds__(256) void gemm_async_wmma_kernel(
    const __bf16* __restrict__ A,   // [M,K] bf16
    const __bf16* __restrict__ Bt,  // [N,K] bf16 (pre-transposed weights)
    __bf16* __restrict__ qw, __bf16* __restrict__ kw, __bf16* __restrict__ vw,
    float* __restrict__ outf) {
  constexpr int K = Cc;
  __shared__ __bf16 As[2][BM * LDA];
  __shared__ __bf16 Bs[2][BN * LDB];

  const int tid  = threadIdx.x;
  const int wave = tid >> 5, lane = tid & 31;
  const int half = (lane >> 4) & 1, l16 = lane & 15;
  const int wm = wave & 3, wn = wave >> 2;  // 4 waves over M, 2 over N
  const int m0 = blockIdx.y * BM, n0 = blockIdx.x * BN;

  const unsigned long long abase = (unsigned long long)A;
  const unsigned long long bbase = (unsigned long long)Bt;
  unsigned asb[2] = { lds_off(&As[0][0]), lds_off(&As[1][0]) };
  unsigned bsb[2] = { lds_off(&Bs[0][0]), lds_off(&Bs[1][0]) };

  auto stage = [&](int buf, int k0) {
#pragma unroll
    for (int i = 0; i < 2; ++i) {           // A tile: 512 x 16B chunks
      int idx = tid + i * 256;
      int row = idx >> 2, c8 = (idx & 3) * 8;
      unsigned lds = asb[buf] + (unsigned)(row * LDA + c8) * 2u;
      unsigned go = (unsigned)(((unsigned long long)(m0 + row) * K + k0 + c8) * 2ull);
      async_copy16(lds, abase, go);
    }
    {                                        // B tile: 256 x 16B chunks
      int row = tid >> 2, c8 = (tid & 3) * 8;
      unsigned lds = bsb[buf] + (unsigned)(row * LDB + c8) * 2u;
      unsigned go = (unsigned)(((unsigned long long)(n0 + row) * K + k0 + c8) * 2ull);
      async_copy16(lds, bbase, go);
    }
  };

  f32x8 acc[2][2];
#pragma unroll
  for (int mt = 0; mt < 2; ++mt)
#pragma unroll
    for (int nt = 0; nt < 2; ++nt)
#pragma unroll
      for (int r = 0; r < 8; ++r) acc[mt][nt][r] = 0.0f;

  stage(0, 0);
  int buf = 0;
  for (int k0 = 0; k0 < K; k0 += BK) {
    const bool more = (k0 + BK) < K;
    if (more) stage(buf ^ 1, k0 + BK);       // prefetch next tile
    if (more) asm volatile("s_wait_asynccnt 3" ::: "memory");
    else      asm volatile("s_wait_asynccnt 0" ::: "memory");
    __syncthreads();

    bf16x16 af[2], bfr[2];
#pragma unroll
    for (int mt = 0; mt < 2; ++mt)
      af[mt] = frag_ld_a(&As[buf][(wm * 32 + mt * 16 + l16) * LDA], half);
#pragma unroll
    for (int nt = 0; nt < 2; ++nt)
      bfr[nt] = frag_ld_b(&Bs[buf][(wn * 32 + nt * 16 + l16) * LDB], half);
#pragma unroll
    for (int mt = 0; mt < 2; ++mt)
#pragma unroll
      for (int nt = 0; nt < 2; ++nt)
        acc[mt][nt] = __builtin_amdgcn_wmma_f32_16x16x32_bf16(
            false, af[mt], false, bfr[nt], (short)0, acc[mt][nt], false, false);
    __syncthreads();
    buf ^= 1;
  }

#pragma unroll
  for (int mt = 0; mt < 2; ++mt) {
#pragma unroll
    for (int nt = 0; nt < 2; ++nt) {
#pragma unroll
      for (int r = 0; r < 8; ++r) {
        int row = m0 + wm * 32 + mt * 16 + half * 8 + r;
        int col = n0 + wn * 32 + nt * 16 + l16;
        float v = acc[mt][nt][r];
        if (MODE == 0) {
          int b = row >> 11, t = row & (Tc - 1);
          int s = col >> 10, hh = (col & 1023) >> 6, e = col & 63;
          size_t dst = ((size_t)(b * Hc + hh) * Tc + t) * HD + e;
          __bf16 bv = (__bf16)v;
          if (s == 0)      qw[dst] = bv;
          else if (s == 1) kw[dst] = bv;
          else             vw[dst] = bv;
        } else {
          outf[(size_t)row * Cc + col] = v;
        }
      }
    }
  }
}

// ---------------------------------------------------------------------------
// Sliding-window attention, flash-style online softmax.
// Block: 128 threads (4 waves), each wave owns 16 query rows (64 per block).
// Per 32-key chunk: 4 WMMA (QK^T over hd=64) + 4 WMMA (PV over hd tiles).
// K chunk staged with async b128; V transposed through ds stores.
// ---------------------------------------------------------------------------
__global__ __launch_bounds__(128) void attn_wmma_kernel(
    const __bf16* __restrict__ qw, const __bf16* __restrict__ kw,
    const __bf16* __restrict__ vw, __bf16* __restrict__ ow,
    const int* __restrict__ win) {
  constexpr float scale = 0.125f;   // hd^-0.5, hd = 64
  __shared__ __bf16 Ks[32 * 64];    // [key][hd]
  __shared__ __bf16 Vs[64 * 32];    // [hd][key] (transposed)
  __shared__ __bf16 Ps[4][16 * 32]; // per-wave P staging [row][key]

  const int tid  = threadIdx.x;
  const int wave = tid >> 5, lane = tid & 31;
  const int half = lane >> 4, l16 = lane & 15;
  const int q0 = blockIdx.x * 64;
  const int bh = blockIdx.y;              // b*H + h
  const int b = bh >> 4, h = bh & 15;
  const int w = win[0];

  const __bf16* kbase = kw + (size_t)bh * Tc * HD;
  const __bf16* vbase = vw + (size_t)bh * Tc * HD;
  const unsigned long long kb64 = (unsigned long long)kbase;
  const unsigned ksb = lds_off(&Ks[0]);

  const __bf16* qrow = qw + ((size_t)bh * Tc + q0 + wave * 16 + l16) * HD;
  bf16x16 qa0 = frag_ld_a(qrow, half);       // hd 0..31
  bf16x16 qa1 = frag_ld_a(qrow + 32, half);  // hd 32..63

  f32x8 o[4];
  float mrow[8], lrow[8];
#pragma unroll
  for (int nt = 0; nt < 4; ++nt)
#pragma unroll
    for (int r = 0; r < 8; ++r) o[nt][r] = 0.0f;
#pragma unroll
  for (int r = 0; r < 8; ++r) { mrow[r] = -3.0e38f; lrow[r] = 0.0f; }

  int kb = q0 - w + 1; if (kb < 0) kb = 0; kb &= ~31;
  for (int kc = kb; kc < q0 + 64; kc += 32) {
    // K chunk: async global->LDS (256 x 16B chunks, 2 per thread)
#pragma unroll
    for (int i = 0; i < 2; ++i) {
      int idx = tid + i * 128;
      int row = idx >> 3, c8 = (idx & 7) * 8;
      unsigned lds = ksb + (unsigned)(row * 64 + c8) * 2u;
      unsigned go = (unsigned)(((unsigned long long)(kc + row) * HD + c8) * 2ull);
      async_copy16(lds, kb64, go);
    }
    // V chunk: load + transpose into Vs[hd][key]
#pragma unroll
    for (int i = 0; i < 2; ++i) {
      int idx = tid + i * 128;
      int row = idx >> 3, c8 = (idx & 7) * 8;
      bf16x8 vv = *reinterpret_cast<const bf16x8*>(
          vbase + (size_t)(kc + row) * HD + c8);
#pragma unroll
      for (int j = 0; j < 8; ++j) Vs[(c8 + j) * 32 + row] = vv[j];
    }
    asm volatile("s_wait_asynccnt 0" ::: "memory");
    __syncthreads();

    // S = Q K^T (two 16-key tiles)
    f32x8 st[2];
#pragma unroll
    for (int t2 = 0; t2 < 2; ++t2) {
      const __bf16* krow = &Ks[(t2 * 16 + l16) * 64];
      bf16x16 kf0 = frag_ld_b(krow, half);        // hd 0..31
      bf16x16 kf1 = frag_ld_b(krow + 32, half);   // hd 32..63
      f32x8 s;
#pragma unroll
      for (int r = 0; r < 8; ++r) s[r] = 0.0f;
      s = __builtin_amdgcn_wmma_f32_16x16x32_bf16(false, qa0, false, kf0,
                                                  (short)0, s, false, false);
      s = __builtin_amdgcn_wmma_f32_16x16x32_bf16(false, qa1, false, kf1,
                                                  (short)0, s, false, false);
      st[t2] = s;
    }

    // mask + scale, row max across the 16 lanes of each half
    float mx[8];
#pragma unroll
    for (int r = 0; r < 8; ++r) {
      int i = q0 + wave * 16 + half * 8 + r;
      int d0 = i - (kc + l16), d1 = i - (kc + 16 + l16);
      float v0 = (d0 >= 0 && d0 < w) ? st[0][r] * scale : -1.0e9f;
      float v1 = (d1 >= 0 && d1 < w) ? st[1][r] * scale : -1.0e9f;
      st[0][r] = v0; st[1][r] = v1;
      mx[r] = fmaxf(v0, v1);
#pragma unroll
      for (int off = 1; off < 16; off <<= 1)
        mx[r] = fmaxf(mx[r], __shfl_xor(mx[r], off, 32));
    }

    // online softmax update + stage P (bf16) for the PV WMMA
    float al[8];
#pragma unroll
    for (int r = 0; r < 8; ++r) {
      float mnew  = fmaxf(mrow[r], mx[r]);
      float alpha = __expf(mrow[r] - mnew);
      float p0 = __expf(st[0][r] - mnew);
      float p1 = __expf(st[1][r] - mnew);
      float rs = p0 + p1;
#pragma unroll
      for (int off = 1; off < 16; off <<= 1) rs += __shfl_xor(rs, off, 32);
      lrow[r] = lrow[r] * alpha + rs;
      mrow[r] = mnew;
      al[r] = alpha;
      Ps[wave][(half * 8 + r) * 32 + l16]      = (__bf16)p0;
      Ps[wave][(half * 8 + r) * 32 + 16 + l16] = (__bf16)p1;
    }
#pragma unroll
    for (int nt = 0; nt < 4; ++nt)
#pragma unroll
      for (int r = 0; r < 8; ++r) o[nt][r] *= al[r];

    // O += P V  (P via per-wave LDS relayout; 4 hd tiles)
    bf16x16 pa = frag_ld_a(&Ps[wave][l16 * 32], half);
#pragma unroll
    for (int nt = 0; nt < 4; ++nt) {
      bf16x16 vb = frag_ld_b(&Vs[(nt * 16 + l16) * 32], half);
      o[nt] = __builtin_amdgcn_wmma_f32_16x16x32_bf16(
          false, pa, false, vb, (short)0, o[nt], false, false);
    }
    __syncthreads();
  }

  // finalize: O /= l, store bf16 to [B][T][H][hd]
#pragma unroll
  for (int r = 0; r < 8; ++r) {
    float inv = 1.0f / lrow[r];
    int t = q0 + wave * 16 + half * 8 + r;
#pragma unroll
    for (int nt = 0; nt < 4; ++nt) {
      int e = nt * 16 + l16;
      ow[(((size_t)b * Tc + t) * Hc + h) * HD + e] = (__bf16)(o[nt][r] * inv);
    }
  }
}

// ---------------------------------------------------------------------------
extern "C" void kernel_launch(void* const* d_in, const int* in_sizes, int n_in,
                              void* d_out, int out_size, void* d_ws,
                              size_t ws_size, hipStream_t stream) {
  const float* x    = (const float*)d_in[0];   // [B,T,C] f32
  const float* Wqkv = (const float*)d_in[1];   // [C,3C] f32
  const float* Wout = (const float*)d_in[2];   // [C,C] f32
  const int*   win  = (const int*)d_in[3];     // scalar window_size
  float* out = (float*)d_out;                  // [B,T,C] f32

  const size_t per = (size_t)Bc * Hc * Tc * HD;    // 4,194,304 elements
  __bf16* qw  = (__bf16*)d_ws;
  __bf16* kw  = qw + per;
  __bf16* vw  = kw + per;
  __bf16* aw  = vw + per;                          // attn out [B][T][H][hd]
  __bf16* xb  = aw + per;                          // x as bf16 [B*T][C]
  __bf16* wqt = xb + per;                          // W_qkv^T bf16 [3C][C]
  __bf16* wot = wqt + (size_t)3 * Cc * Cc;         // W_out^T bf16 [C][C]

  // precision / layout conversion passes
  cvt_f32_bf16_kernel<<<(Bc * Tc * Cc / 4 + 255) / 256, 256, 0, stream>>>(
      x, xb, Bc * Tc * Cc / 4);
  cvt_transpose_kernel<<<(3 * Cc * Cc + 255) / 256, 256, 0, stream>>>(
      Wqkv, wqt, 3 * Cc, 3 * Cc * Cc);
  cvt_transpose_kernel<<<(Cc * Cc + 255) / 256, 256, 0, stream>>>(
      Wout, wot, Cc, Cc * Cc);

  dim3 g1((3 * Cc) / BN, (Bc * Tc) / BM);          // 48 x 32
  gemm_async_wmma_kernel<0><<<g1, 256, 0, stream>>>(xb, wqt, qw, kw, vw,
                                                    nullptr);

  dim3 g2(Tc / 64, Bc * Hc);                       // 32 x 32
  attn_wmma_kernel<<<g2, 128, 0, stream>>>(qw, kw, vw, aw, win);

  dim3 g3(Cc / BN, (Bc * Tc) / BM);                // 16 x 32
  gemm_async_wmma_kernel<1><<<g3, 256, 0, stream>>>(aw, wot, nullptr, nullptr,
                                                    nullptr, out);
}
